// ODLUE_13941463843221
// MI455X (gfx1250) — matrix operand order repaired and checked by reading, not compiled, not run on hardware
//
#include <hip/hip_runtime.h>
#include <hip/hip_bf16.h>

typedef __attribute__((ext_vector_type(16))) _Float16 v16h;
typedef __attribute__((ext_vector_type(8)))  _Float16 v8h;
typedef __attribute__((ext_vector_type(2)))  _Float16 v2h;
typedef __attribute__((ext_vector_type(8)))  float    v8f;

namespace {
constexpr int DH  = 768;     // 32 days * 24 hours
constexpr int NL  = 2000;    // links
constexpr int NP  = 20000;   // paths
constexpr int NO  = 4000;    // OD pairs
constexpr int NF  = 8;       // features
constexpr int KP  = 2016;    // links padded to multiple of 32 (WMMA K for GEMM1)
constexpr int NP2 = 20096;   // paths padded to multiple of 128 (GEMM1 N)
constexpr int NL2 = 2048;    // links padded to multiple of 128 (GEMM2 N)
}

__device__ __forceinline__ unsigned odlue_encf(float f) {
  unsigned u = __float_as_uint(f);
  return (u & 0x80000000u) ? ~u : (u | 0x80000000u);
}
__device__ __forceinline__ float odlue_decf(unsigned u) {
  return __uint_as_float((u & 0x80000000u) ? (u & 0x7fffffffu) : ~u);
}

// ---------------- small prep kernels ----------------

// psc * log(path_size_factors), zero-padded to NP2 columns
__global__ void odlue_psc(const float* __restrict__ psf, const float* __restrict__ psc,
                          float* __restrict__ pscT) {
  int p = blockIdx.x * blockDim.x + threadIdx.x;
  if (p < NP2) pscT[p] = (p < NP) ? psc[0] * __logf(psf[p]) : 0.0f;
}

__global__ void odlue_zero(unsigned* __restrict__ buf, int n) {
  int i = blockIdx.x * blockDim.x + threadIdx.x;
  if (i < n) buf[i] = 0u;
}

// V[dh][l] = X[dh,l,:] . theta + theta_links[l]  -> f16, K padded with zeros
__global__ void odlue_linkutil(const float* __restrict__ X, const float* __restrict__ theta,
                               const float* __restrict__ thetaL, _Float16* __restrict__ Vf16) {
  int idx = blockIdx.x * blockDim.x + threadIdx.x;
  if (idx >= DH * KP) return;
  int dh = idx / KP;
  int l  = idx - dh * KP;
  float v = 0.0f;
  if (l < NL) {
    const float* x = X + ((size_t)dh * NL + l) * NF;
    float s = thetaL[l];
#pragma unroll
    for (int f = 0; f < NF; ++f) s += x[f] * theta[f];
    v = s;
  }
  Vf16[idx] = (_Float16)v;
}

// Dmat (NL x NP, f32) -> Df16 (KP x NP2, f16, pad rows/cols zero)
//                     -> DTf16 (NP x NL2, f16, pad cols zero), LDS-tiled transpose.
// Grid: (NP2/32, NL2/32), block (32, 8).
__global__ void odlue_dconv(const float* __restrict__ D, _Float16* __restrict__ Df16,
                            _Float16* __restrict__ DTf16) {
  __shared__ _Float16 tile[32][33];
  const int p0 = blockIdx.x * 32;
  const int l0 = blockIdx.y * 32;
  const int tx = threadIdx.x, ty = threadIdx.y;
#pragma unroll
  for (int i = 0; i < 4; ++i) {
    const int l = l0 + ty + i * 8;
    const int p = p0 + tx;
    _Float16 v = (_Float16)0.0f;
    if (l < NL && p < NP) v = (_Float16)D[(size_t)l * NP + p];
    if (l < KP) Df16[(size_t)l * NP2 + p] = v;
    tile[ty + i * 8][tx] = v;
  }
  __syncthreads();
#pragma unroll
  for (int i = 0; i < 4; ++i) {
    const int p = p0 + ty + i * 8;
    const int l = l0 + tx;
    if (p < NP) DTf16[(size_t)p * NL2 + l] = tile[tx][ty + i * 8];
  }
}

// ---------------- WMMA GEMM: C(MxN,f32) = A(MxK,f16) * B(KxNpad,f16) [+ colBias] ----
// Block tile 128x128, BK=32, 256 threads = 8 waves in 2(M)x4(N); wave tile 64x32 = 4x2 frags.
// Double-buffered LDS pipeline. Requires M%128==0, K%32==0 (A padded), B has >= gridX*128
// zero-padded columns; C stores guarded by gn < N only.
__launch_bounds__(256)
__global__ void odlue_wmma_gemm(const _Float16* __restrict__ A, int lda,
                                const _Float16* __restrict__ B, int ldb,
                                float* __restrict__ C, int ldc,
                                const float* __restrict__ colBias,
                                int N, int kSteps) {
  constexpr int LDS_S = 40;  // halves, padded stride
  __shared__ __align__(16) _Float16 ldsA[2][128 * LDS_S];  // [buf][row][k]
  __shared__ __align__(16) _Float16 ldsB[2][128 * LDS_S];  // [buf][n][k] (transposed)
  const int tid = threadIdx.x;
  const int lid = tid & 31;
  const int wv  = tid >> 5;
  const int wM  = wv >> 2;   // 0..1
  const int wN  = wv & 3;    // 0..3
  const long mBlock = (long)blockIdx.y * 128;
  const int  nBlock = blockIdx.x * 128;

  // ---- thread-constant staging addresses ----
  // A: two 8-half chunks: rows (tid>>2) and (tid>>2)+64, column seg (tid&3)*8
  const int aRow = tid >> 2;
  const int aSeg = (tid & 3) * 8;
  const _Float16* aG0 = A + (mBlock + aRow) * (long)lda + aSeg;
  const _Float16* aG1 = aG0 + 64 * (long)lda;
  const int aS0 = aRow * LDS_S + aSeg;
  const int aS1 = aS0 + 64 * LDS_S;
  // B: 2(k) x 8(n) chunk: k rows 2*(tid>>4), 2*(tid>>4)+1, columns (tid&15)*8..+7
  const int kPair = (tid >> 4) * 2;
  const int nLoc  = (tid & 15) * 8;
  const _Float16* bG0 = B + (long)kPair * ldb + nBlock + nLoc;
  const _Float16* bG1 = bG0 + ldb;
  const int bS = nLoc * LDS_S + kPair;

  // ---- thread-constant fragment LDS offsets ----
  int aOff[4], bOff[2];
#pragma unroll
  for (int fm = 0; fm < 4; ++fm)
    aOff[fm] = (wM * 64 + fm * 16 + (lid & 15)) * LDS_S + ((lid < 16) ? 0 : 8);
#pragma unroll
  for (int fn = 0; fn < 2; ++fn)
    bOff[fn] = (wN * 32 + fn * 16 + (lid & 15)) * LDS_S + ((lid < 16) ? 0 : 8);

  v8f acc[4][2];
#pragma unroll
  for (int i = 0; i < 4; ++i)
#pragma unroll
    for (int j = 0; j < 2; ++j)
#pragma unroll
      for (int r = 0; r < 8; ++r) acc[i][j][r] = 0.0f;

  // ---- prologue: stage tile 0 into buffer 0 ----
  {
    v8h a0 = *(const v8h*)aG0;
    v8h a1 = *(const v8h*)aG1;
    v8h b0 = *(const v8h*)bG0;
    v8h b1 = *(const v8h*)bG1;
    *(v8h*)(ldsA[0] + aS0) = a0;
    *(v8h*)(ldsA[0] + aS1) = a1;
#pragma unroll
    for (int j = 0; j < 8; ++j) {
      v2h pk; pk[0] = b0[j]; pk[1] = b1[j];
      *(v2h*)(ldsB[0] + bS + j * LDS_S) = pk;
    }
  }
  __syncthreads();

  int cur = 0;
  for (int ks = 0; ks < kSteps; ++ks) {
    const bool hasNext = (ks + 1 < kSteps);
    v8h na0, na1, nb0, nb1;
    if (hasNext) {  // prefetch next tile from global into registers
      const long ka = (long)(ks + 1) * 32;
      na0 = *(const v8h*)(aG0 + ka);
      na1 = *(const v8h*)(aG1 + ka);
      nb0 = *(const v8h*)(bG0 + ka * ldb);
      nb1 = *(const v8h*)(bG1 + ka * ldb);
    }

    // ---- fragments from current buffer ----
    const _Float16* lA = ldsA[cur];
    const _Float16* lB = ldsB[cur];
    v16h bf[2], af[4];
#pragma unroll
    for (int fn = 0; fn < 2; ++fn) {
      v8h lo = *(const v8h*)(lB + bOff[fn]);
      v8h hi = *(const v8h*)(lB + bOff[fn] + 16);
      bf[fn] = __builtin_shufflevector(lo, hi, 0,1,2,3,4,5,6,7,8,9,10,11,12,13,14,15);
    }
#pragma unroll
    for (int fm = 0; fm < 4; ++fm) {
      v8h lo = *(const v8h*)(lA + aOff[fm]);
      v8h hi = *(const v8h*)(lA + aOff[fm] + 16);
      af[fm] = __builtin_shufflevector(lo, hi, 0,1,2,3,4,5,6,7,8,9,10,11,12,13,14,15);
    }
#pragma unroll
    for (int fm = 0; fm < 4; ++fm)
#pragma unroll
      for (int fn = 0; fn < 2; ++fn)
        acc[fm][fn] = __builtin_amdgcn_wmma_f32_16x16x32_f16(
            false, af[fm], false, bf[fn], (short)0, acc[fm][fn], false, false);

    if (hasNext) {  // store prefetched tile into the other buffer
      _Float16* nA = ldsA[cur ^ 1];
      _Float16* nB = ldsB[cur ^ 1];
      *(v8h*)(nA + aS0) = na0;
      *(v8h*)(nA + aS1) = na1;
#pragma unroll
      for (int j = 0; j < 8; ++j) {
        v2h pk; pk[0] = nb0[j]; pk[1] = nb1[j];
        *(v2h*)(nB + bS + j * LDS_S) = pk;
      }
      __syncthreads();
      cur ^= 1;
    }
  }

  // ---- epilogue: C/D layout -> VGPR r = (M = r + (lane>=16 ? 8 : 0), N = lane&15)
  const int rowAdd = (lid < 16) ? 0 : 8;
  const int nCol = lid & 15;
#pragma unroll
  for (int fm = 0; fm < 4; ++fm) {
#pragma unroll
    for (int fn = 0; fn < 2; ++fn) {
      const int gn = nBlock + wN * 32 + fn * 16 + nCol;
      if (gn < N) {
        const float bias = colBias ? colBias[gn] : 0.0f;
#pragma unroll
        for (int r = 0; r < 8; ++r) {
          const long gm = mBlock + wM * 64 + fm * 16 + r + rowAdd;
          C[gm * (long)ldc + gn] = acc[fm][fn][r] + bias;
        }
      }
    }
  }
}

// ---------------- segment softmax + flows (vf/Ff16 have row stride NP2) ------------

__global__ void odlue_segmax(const float* __restrict__ vf, const int* __restrict__ od,
                             unsigned* __restrict__ segmx) {
  int idx = blockIdx.x * blockDim.x + threadIdx.x;
  if (idx >= DH * NP) return;
  int dh = idx / NP;
  int p  = idx - dh * NP;
  atomicMax(&segmx[od[p] * DH + dh], odlue_encf(vf[(size_t)dh * NP2 + p]));
}

__global__ void odlue_expsum(float* __restrict__ vf, const int* __restrict__ od,
                             const unsigned* __restrict__ segmx, float* __restrict__ segsm) {
  int idx = blockIdx.x * blockDim.x + threadIdx.x;
  if (idx >= DH * NP) return;
  int dh = idx / NP;
  int p  = idx - dh * NP;
  int o  = od[p];
  float m = odlue_decf(segmx[o * DH + dh]);
  float e = __expf(vf[(size_t)dh * NP2 + p] - m);
  vf[(size_t)dh * NP2 + p] = e;
  atomicAdd(&segsm[o * DH + dh], e);
}

__global__ void odlue_flow(const float* __restrict__ vf, const int* __restrict__ od,
                           const float* __restrict__ segsm, const float* __restrict__ q_sqrt,
                           _Float16* __restrict__ Ff16) {
  int idx = blockIdx.x * blockDim.x + threadIdx.x;
  if (idx >= DH * NP) return;
  int dh = idx / NP;
  int p  = idx - dh * NP;
  int o  = od[p];
  float q = q_sqrt[o];
  q *= q;
  Ff16[(size_t)dh * NP2 + p] = (_Float16)(q * vf[(size_t)dh * NP2 + p] / segsm[o * DH + dh]);
}

// ---------------- launcher ----------------

extern "C" void kernel_launch(void* const* d_in, const int* in_sizes, int n_in,
                              void* d_out, int out_size, void* d_ws, size_t ws_size,
                              hipStream_t stream) {
  (void)in_sizes; (void)n_in; (void)out_size; (void)ws_size;
  const float* X      = (const float*)d_in[0];
  const float* theta  = (const float*)d_in[1];
  const float* thetaL = (const float*)d_in[2];
  const float* q_sqrt = (const float*)d_in[3];
  const float* psc    = (const float*)d_in[4];
  const float* psf    = (const float*)d_in[5];
  const float* Dmat   = (const float*)d_in[6];
  const int*   od_ids = (const int*)d_in[7];
  float* out = (float*)d_out;

  char* ws = (char*)d_ws;
  auto take = [&](size_t bytes) {
    char* p = ws;
    ws += (bytes + 255) & ~(size_t)255;
    return p;
  };
  _Float16* Vf16  = (_Float16*)take((size_t)DH * KP * 2);    // padded link utilities (f16)
  _Float16* Df16  = (_Float16*)take((size_t)KP * NP2 * 2);   // D in f16, K- and N-padded
  _Float16* DTf16 = (_Float16*)take((size_t)NP * NL2 * 2);   // D^T in f16, N-padded
  float*    vf    = (float*)take((size_t)DH * NP2 * 4);      // path utilities / exp
  unsigned* segmx = (unsigned*)take((size_t)NO * DH * 4);    // encoded per-OD max
  float*    segsm = (float*)take((size_t)NO * DH * 4);       // per-OD exp sums
  _Float16* Ff16  = (_Float16*)take((size_t)DH * NP2 * 2);   // path flows (f16)
  float*    pscT  = (float*)take((size_t)NP2 * 4);           // psc * log(psf), padded

  odlue_psc<<<(NP2 + 255) / 256, 256, 0, stream>>>(psf, psc, pscT);
  odlue_linkutil<<<(DH * KP + 255) / 256, 256, 0, stream>>>(X, theta, thetaL, Vf16);
  odlue_dconv<<<dim3(NP2 / 32, NL2 / 32), dim3(32, 8), 0, stream>>>(Dmat, Df16, DTf16);
  odlue_zero<<<(NO * DH + 255) / 256, 256, 0, stream>>>(segmx, NO * DH);
  odlue_zero<<<(NO * DH + 255) / 256, 256, 0, stream>>>((unsigned*)segsm, NO * DH);

  // GEMM1: vf(DH x NP2) = V(DH x KP) * Df16(KP x NP2) + pscT broadcast over columns
  odlue_wmma_gemm<<<dim3(NP2 / 128, DH / 128), 256, 0, stream>>>(
      Vf16, KP, Df16, NP2, vf, NP2, pscT, NP2, KP / 32);

  odlue_segmax<<<(DH * NP + 255) / 256, 256, 0, stream>>>(vf, od_ids, segmx);
  odlue_expsum<<<(DH * NP + 255) / 256, 256, 0, stream>>>(vf, od_ids, segmx, segsm);
  odlue_flow<<<(DH * NP + 255) / 256, 256, 0, stream>>>(vf, od_ids, segsm, q_sqrt, Ff16);

  // GEMM2: out(DH x NL) = F(DH x NP) * D^T(NP x NL2), stores guarded to NL
  odlue_wmma_gemm<<<dim3(NL2 / 128, DH / 128), 256, 0, stream>>>(
      Ff16, NP2, DTf16, NL2, out, NL, nullptr, NL, NP / 32);
}
